// EarthAttention3D_27968827031824
// MI455X (gfx1250) — matrix operand order repaired and checked by reading, not compiled, not run on hardware
//
#include <hip/hip_runtime.h>
#include <hip/hip_bf16.h>

// ---------------------------------------------------------------------------
// EarthAttention3D for MI455X (gfx1250), wave32 + WMMA bf16 (f32 accumulate).
//
// Shapes: B_=30, nW=32, N=144, DIM=192, H=6, hd=32.  M = 138240 token rows.
// ~56 GFLOP total vs ~300MB HBM traffic @23.3TB/s -> keep attention scores
// resident in LDS per (b,w,h) block, bf16 intermediates, WMMA 16x16x32 bf16.
// Workspace: Q,K,V,attn-out as bf16, 4 * 26,542,080 * 2B ~= 212 MB.
// ---------------------------------------------------------------------------

typedef __attribute__((ext_vector_type(16))) __bf16 v16bf;
typedef __attribute__((ext_vector_type(8)))  float  v8f;

union FragBF {
    uint4          q[2];   // two 16B chunks (8 bf16 each)
    unsigned short u[16];
    v16bf          v;
};

__device__ __forceinline__ unsigned short f2bf(float f) {
    unsigned int x = __builtin_bit_cast(unsigned int, f);
    unsigned int r = x + 0x7fffu + ((x >> 16) & 1u);   // round-to-nearest-even
    return (unsigned short)(r >> 16);
}

#define WMMA_BF16(A, B, C) \
    __builtin_amdgcn_wmma_f32_16x16x32_bf16(false, (A), false, (B), (short)0, (C), false, false)

// ---------------------------------------------------------------------------
// Kernel 1: QKV projection.  out[m, c] = x[m,:] @ qkv_w[c,:] + qkv_b[c]
// Block tile 64(M) x 64(C), K=192 staged in LDS as bf16 (stride 200 -> 16B
// aligned rows, bank-conflict friendly).  8 waves: 2(M-groups) x 4(N-groups),
// each wave owns a 32x16 output strip = 2 WMMA accumulators.
// Results scattered to per-(b,w,h) token-major bf16 tiles; q pre-scaled.
// ---------------------------------------------------------------------------
__global__ __launch_bounds__(256) void qkv_kernel(
    const float* __restrict__ x, const float* __restrict__ qkv_w,
    const float* __restrict__ qkv_b,
    unsigned short* __restrict__ qb, unsigned short* __restrict__ kb,
    unsigned short* __restrict__ vb)
{
    constexpr int LD = 200;                     // ushorts per LDS row (400B)
    extern __shared__ char smemRaw[];
    unsigned short* sA = (unsigned short*)smemRaw;   // 64 x 200
    unsigned short* sB = sA + 64 * LD;               // 64 x 200

    const int tid  = threadIdx.x;
    const int mBlk = blockIdx.x * 64;
    const int nBlk = blockIdx.y * 64;

    for (int idx = tid; idx < 64 * 192; idx += 256) {
        int r = idx / 192, c = idx % 192;
        sA[r * LD + c] = f2bf(x[(size_t)(mBlk + r) * 192 + c]);
        sB[r * LD + c] = f2bf(qkv_w[(size_t)(nBlk + r) * 192 + c]);
    }
    __syncthreads();

    const int wave  = tid >> 5, lane = tid & 31;
    const int l15   = lane & 15, lhalf = lane >> 4;
    const int waveM = (wave & 1) * 32;
    const int waveN = (wave >> 1) * 16;

    v8f acc0 = {}, acc1 = {};
    for (int kbase = 0; kbase < 192; kbase += 32) {
        FragBF bf;
        {   // B fragment: column-per-lane, K {0..7,16..23} / {8..15,24..31}
            const unsigned short* p = sB + (waveN + l15) * LD + kbase + (lhalf ? 8 : 0);
            bf.q[0] = *(const uint4*)p;
            bf.q[1] = *(const uint4*)(p + 16);
        }
        FragBF a0, a1;
        {   // A fragment: row-per-lane, K {0..7,16..23} / {8..15,24..31}
            const unsigned short* p = sA + (waveM + l15) * LD + kbase + (lhalf ? 8 : 0);
            a0.q[0] = *(const uint4*)p;
            a0.q[1] = *(const uint4*)(p + 16);
        }
        {
            const unsigned short* p = sA + (waveM + 16 + l15) * LD + kbase + (lhalf ? 8 : 0);
            a1.q[0] = *(const uint4*)p;
            a1.q[1] = *(const uint4*)(p + 16);
        }
        acc0 = WMMA_BF16(a0.v, bf.v, acc0);
        acc1 = WMMA_BF16(a1.v, bf.v, acc1);
    }

    // Scatter D (lane<16: rows r0..r0+7 col=lane; lane>=16: rows +8..+15)
    const int col = nBlk + waveN + l15;
    const int s   = col / 192;            // 0=q, 1=k, 2=v
    const int rem = col % 192;
    const int hh  = rem >> 5, d = rem & 31;
    const float bias  = qkv_b[col];
    const float scale = (s == 0) ? 0.17677669529663687f : 1.0f;  // hd^-0.5
    unsigned short* dst = (s == 0) ? qb : (s == 1) ? kb : vb;

    #pragma unroll
    for (int i = 0; i < 2; ++i) {
        const v8f acc = i ? acc1 : acc0;
        const int rowBase = mBlk + waveM + i * 16 + (lhalf ? 8 : 0);
        #pragma unroll
        for (int r = 0; r < 8; ++r) {
            int row = rowBase + r;
            int bw = row / 144, tok = row % 144;
            float val = (acc[r] + bias) * scale;
            dst[(((size_t)bw * 6 + hh) * 144 + tok) * 32 + d] = f2bf(val);
        }
    }
}

// ---------------------------------------------------------------------------
// Kernel 2: attention per (b, window, head).  5760 blocks x 256 threads.
// K (144x32) + V^T (32x160+pad) staged once; queries in 3 chunks of 48 rows.
// S = Q@K^T (one WMMA per 16x16 tile, K=32) -> f32 in LDS; +bias gather,
// +mask, wave32 softmax; O = P@V with K padded to 160 (pads zeroed both
// sides so WMMA NaN rules can't bite).
// ---------------------------------------------------------------------------
__global__ __launch_bounds__(256) void attn_kernel(
    const unsigned short* __restrict__ qbuf, const unsigned short* __restrict__ kbuf,
    const unsigned short* __restrict__ vbuf, const float* __restrict__ mask,
    const float* __restrict__ bias_table, const int* __restrict__ pos_index,
    unsigned short* __restrict__ aout)
{
    constexpr int LQK = 40;    // Q/K row stride (80B, 16B aligned)
    constexpr int LVT = 168;   // V^T row stride (336B)
    constexpr int LS  = 168;   // score row stride (672B)
    extern __shared__ char smemRaw[];
    unsigned short* sK  = (unsigned short*)smemRaw;        // 144*40  = 11520B
    unsigned short* sVt = sK + 144 * LQK;                  // 32*168  = 10752B
    unsigned short* sQ  = sVt + 32 * LVT;                  // 48*40   =  3840B
    float*          sS  = (float*)(sQ + 48 * LQK);         // 48*168  = 32256B

    const int tid = threadIdx.x;
    const int bwh = blockIdx.x;
    const int h   = bwh % 6;
    const int bw  = bwh / 6;              // b*32 + wI
    const int wI  = bw % 32;
    const size_t base = (size_t)bwh * (144 * 32);

    // Stage K and transposed V
    for (int idx = tid; idx < 144 * 32; idx += 256) {
        int tok = idx >> 5, d = idx & 31;
        sK[tok * LQK + d]  = kbuf[base + idx];
        sVt[d * LVT + tok] = vbuf[base + idx];
    }
    for (int idx = tid; idx < 32 * 24; idx += 256) {       // zero V^T K-pad 144..167
        int d = idx / 24, c = idx % 24;
        sVt[d * LVT + 144 + c] = 0;
    }
    __syncthreads();

    const int wave = tid >> 5, lane = tid & 31;
    const int l15  = lane & 15, lhalf = lane >> 4;
    const size_t maskBase = (size_t)bw * 144 * 144;
    const int    biasWH   = wI * 6 + h;
    const size_t outBase  = (size_t)bw * (144 * 192) + (size_t)h * 32;

    for (int chunk = 0; chunk < 3; ++chunk) {
        const int rowBase = chunk * 48;
        for (int idx = tid; idx < 48 * 32; idx += 256) {
            int tok = idx >> 5, d = idx & 31;
            sQ[tok * LQK + d] = qbuf[base + (size_t)(rowBase + tok) * 32 + d];
        }
        __syncthreads();

        // ---- S = Q @ K^T : 3 x 9 tiles of 16x16, single K=32 step ----
        for (int t = wave; t < 27; t += 8) {
            int tm = t / 9, tn = t % 9;
            FragBF a, bf;
            { const unsigned short* p = sQ + (tm * 16 + l15) * LQK + (lhalf ? 8 : 0);
              a.q[0] = *(const uint4*)p;  a.q[1] = *(const uint4*)(p + 16); }
            { const unsigned short* p = sK + (tn * 16 + l15) * LQK + (lhalf ? 8 : 0);
              bf.q[0] = *(const uint4*)p; bf.q[1] = *(const uint4*)(p + 16); }
            v8f acc = {};
            acc = WMMA_BF16(a.v, bf.v, acc);
            int c  = tn * 16 + l15;
            int r0 = tm * 16 + (lhalf ? 8 : 0);
            #pragma unroll
            for (int r = 0; r < 8; ++r) sS[(r0 + r) * LS + c] = acc[r];
        }
        __syncthreads();

        // ---- bias + mask + softmax (one wave per row, lanes stride cols) ----
        for (int lr = wave; lr < 48; lr += 8) {
            const int row = rowBase + lr;
            float vals[5];
            float mx = -3.0e38f;
            #pragma unroll
            for (int it = 0; it < 5; ++it) {
                int c = lane + it * 32;
                float vv = -3.0e38f;
                if (c < 144) {
                    vv = sS[lr * LS + c]
                       + bias_table[(size_t)pos_index[row * 144 + c] * 192 + biasWH]
                       + mask[maskBase + (size_t)row * 144 + c];
                }
                vals[it] = vv;
                mx = fmaxf(mx, vv);
            }
            #pragma unroll
            for (int off = 16; off; off >>= 1) mx = fmaxf(mx, __shfl_xor(mx, off, 32));
            float sum = 0.f;
            #pragma unroll
            for (int it = 0; it < 5; ++it) {
                int c = lane + it * 32;
                float e = (c < 144) ? __expf(vals[it] - mx) : 0.f;
                vals[it] = e;
                sum += e;
            }
            #pragma unroll
            for (int off = 16; off; off >>= 1) sum += __shfl_xor(sum, off, 32);
            float inv = 1.0f / sum;
            #pragma unroll
            for (int it = 0; it < 5; ++it) {
                int c = lane + it * 32;
                if (c < 144) sS[lr * LS + c] = vals[it] * inv;
            }
            int c = 144 + lane;                       // zero P K-pad 144..159
            if (c < 160) sS[lr * LS + c] = 0.f;
        }
        __syncthreads();

        // ---- O = P @ V : 3 x 2 tiles, K=160 (5 WMMA steps) ----
        for (int t = wave; t < 6; t += 8) {
            int tm = t >> 1, tn = t & 1;
            v8f acc = {};
            for (int kbase = 0; kbase < 160; kbase += 32) {
                FragBF a, bf;
                {   // convert f32 probabilities -> bf16 A fragment on the fly
                    const float* p = sS + (tm * 16 + l15) * LS + kbase + (lhalf ? 8 : 0);
                    #pragma unroll
                    for (int e = 0; e < 8; ++e) a.u[e]     = f2bf(p[e]);
                    #pragma unroll
                    for (int e = 0; e < 8; ++e) a.u[8 + e] = f2bf(p[16 + e]);
                }
                { const unsigned short* p = sVt + (tn * 16 + l15) * LVT + kbase + (lhalf ? 8 : 0);
                  bf.q[0] = *(const uint4*)p; bf.q[1] = *(const uint4*)(p + 16); }
                acc = WMMA_BF16(a.v, bf.v, acc);
            }
            int dcol = tn * 16 + l15;
            int r0   = rowBase + tm * 16 + (lhalf ? 8 : 0);
            #pragma unroll
            for (int r = 0; r < 8; ++r)
                aout[outBase + (size_t)(r0 + r) * 192 + dcol] = f2bf(acc[r]);
        }
        __syncthreads();
    }
}

// ---------------------------------------------------------------------------
// Kernel 3: output projection.  out[m,n] = attn[m,:] @ proj_w[n,:] + proj_b[n]
// Same 64x64 WMMA tiling as kernel 1; A already bf16 in workspace.
// ---------------------------------------------------------------------------
__global__ __launch_bounds__(256) void proj_kernel(
    const unsigned short* __restrict__ ain, const float* __restrict__ proj_w,
    const float* __restrict__ proj_b, float* __restrict__ out)
{
    constexpr int LD = 200;
    extern __shared__ char smemRaw[];
    unsigned short* sA = (unsigned short*)smemRaw;
    unsigned short* sB = sA + 64 * LD;

    const int tid  = threadIdx.x;
    const int mBlk = blockIdx.x * 64;
    const int nBlk = blockIdx.y * 64;

    for (int idx = tid; idx < 64 * 192; idx += 256) {
        int r = idx / 192, c = idx % 192;
        sA[r * LD + c] = ain[(size_t)(mBlk + r) * 192 + c];
        sB[r * LD + c] = f2bf(proj_w[(size_t)(nBlk + r) * 192 + c]);
    }
    __syncthreads();

    const int wave  = tid >> 5, lane = tid & 31;
    const int l15   = lane & 15, lhalf = lane >> 4;
    const int waveM = (wave & 1) * 32;
    const int waveN = (wave >> 1) * 16;

    v8f acc0 = {}, acc1 = {};
    for (int kbase = 0; kbase < 192; kbase += 32) {
        FragBF bf;
        { const unsigned short* p = sB + (waveN + l15) * LD + kbase + (lhalf ? 8 : 0);
          bf.q[0] = *(const uint4*)p; bf.q[1] = *(const uint4*)(p + 16); }
        FragBF a0, a1;
        { const unsigned short* p = sA + (waveM + l15) * LD + kbase + (lhalf ? 8 : 0);
          a0.q[0] = *(const uint4*)p; a0.q[1] = *(const uint4*)(p + 16); }
        { const unsigned short* p = sA + (waveM + 16 + l15) * LD + kbase + (lhalf ? 8 : 0);
          a1.q[0] = *(const uint4*)p; a1.q[1] = *(const uint4*)(p + 16); }
        acc0 = WMMA_BF16(a0.v, bf.v, acc0);
        acc1 = WMMA_BF16(a1.v, bf.v, acc1);
    }

    const int col  = nBlk + waveN + l15;
    const float bs = proj_b[col];
    #pragma unroll
    for (int i = 0; i < 2; ++i) {
        const v8f acc = i ? acc1 : acc0;
        const int rowBase = mBlk + waveM + i * 16 + (lhalf ? 8 : 0);
        #pragma unroll
        for (int r = 0; r < 8; ++r)
            out[(size_t)(rowBase + r) * 192 + col] = acc[r] + bs;
    }
}

// ---------------------------------------------------------------------------
extern "C" void kernel_launch(void* const* d_in, const int* in_sizes, int n_in,
                              void* d_out, int out_size, void* d_ws, size_t ws_size,
                              hipStream_t stream) {
    (void)in_sizes; (void)n_in; (void)out_size; (void)ws_size;

    const float* x          = (const float*)d_in[0];
    const float* mask       = (const float*)d_in[1];
    const float* qkv_w      = (const float*)d_in[2];
    const float* qkv_b      = (const float*)d_in[3];
    const float* proj_w     = (const float*)d_in[4];
    const float* proj_b     = (const float*)d_in[5];
    const float* bias_table = (const float*)d_in[6];
    const int*   pos_index  = (const int*)d_in[7];
    float* out = (float*)d_out;

    // Workspace: Q,K,V,attn-out as bf16 (ushort). 4 * 26,542,080 * 2B = 212MB.
    const size_t EQ = (size_t)5760 * 144 * 32;   // 30*32*6 heads * 144 tok * 32
    unsigned short* ws = (unsigned short*)d_ws;
    unsigned short* qb = ws;
    unsigned short* kb = ws + EQ;
    unsigned short* vb = ws + 2 * EQ;
    unsigned short* ab = ws + 3 * EQ;

    const size_t shGemm = (size_t)2 * 64 * 200 * sizeof(unsigned short); // 51200B
    const size_t shAttn = 11520 + 10752 + 3840 + 32256;                  // 58368B

    qkv_kernel<<<dim3(2160, 9), dim3(256), shGemm, stream>>>(x, qkv_w, qkv_b, qb, kb, vb);
    attn_kernel<<<dim3(5760), dim3(256), shAttn, stream>>>(qb, kb, vb, mask,
                                                           bias_table, pos_index, ab);
    proj_kernel<<<dim3(2160, 3), dim3(256), shGemm, stream>>>(ab, proj_w, proj_b, out);
}
